// FastCNNBlock3DConv2PartDropout_41523743817982
// MI455X (gfx1250) — matrix operand order repaired and checked by reading, not compiled, not run on hardware
//
#include <hip/hip_runtime.h>
#include <hip/hip_bf16.h>

// ---------- CDNA5 WMMA / TDM types ----------
typedef __attribute__((ext_vector_type(16))) __bf16       v16bf;
typedef __attribute__((ext_vector_type(8)))  float        v8f;
typedef __attribute__((ext_vector_type(4)))  unsigned int v4u;
typedef __attribute__((ext_vector_type(8)))  int          v8i;
typedef __attribute__((ext_vector_type(4)))  int          v4i;

// ---------- problem constants ----------
constexpr int  D3   = 80;
constexpr long VOX  = (long)D3 * D3 * D3;     // 512000
constexpr int  NPTS = 200000;
constexpr int  TILE_X = 32;
constexpr int  SLAB_W = TILE_X + 2;           // 34

// A-fragment K mapping for 16-bit WMMA A (16x32), contiguous order matching B:
// lanes 0-15: K = {0..15}; lanes 16-31: K = {16..31}
__device__ __host__ inline int kmapA(int lane, int j) {
    return ((lane < 16) ? 0 : 16) + j;
}

// =====================================================================
// TDM: 1-D contiguous tile, global -> LDS. data_size = 8 bytes.
// D# bit layout per CDNA5 ISA 8.3/8.4. 6-arg builtin (clang-23 flavor).
// =====================================================================
__device__ inline void tdm_load_1d(const void* gptr, void* lptr,
                                   unsigned int tile_bytes,
                                   unsigned long long total_bytes) {
    unsigned long long ga = (unsigned long long)gptr;
    unsigned int lds = (unsigned int)(unsigned long long)lptr; // addr[31:0] = LDS offset
    unsigned int tile_e = tile_bytes >> 3;                 // 8B units (<= 65535)
    unsigned long long tot_e = total_bytes >> 3;           // tensor_dim0

    v4u g0;
    g0[0] = 1u;                                            // count=1, flags=0
    g0[1] = lds;                                           // lds_addr
    g0[2] = (unsigned int)(ga & 0xFFFFFFFFu);              // global_addr[31:0]
    g0[3] = (unsigned int)((ga >> 32) & 0x01FFFFFFu) | 0x80000000u; // [56:32] | type=2

    v8i g1;
    g1[0] = (int)(3u << 16);                               // data_size=3 (8B); wg_mask=0
    g1[1] = (int)((tot_e & 0xFFFFu) << 16);                // tensor_dim0[15:0] @ bits[63:48]
    g1[2] = (int)((tot_e >> 16) & 0xFFFFu) | (1 << 16);    // tensor_dim0[31:16] | tensor_dim1=1
    g1[3] = (int)(tile_e << 16);                           // tile_dim0 @ bits[127:112]
    g1[4] = 0;                                             // tile_dim1=0, tile_dim2=0 (1-D)
    g1[5] = (int)(tot_e & 0xFFFFFFFFu);                    // tensor_dim0_stride lo
    g1[6] = (int)((tot_e >> 32) & 0xFFFFu);                // stride hi; dim1_stride lo = 0
    g1[7] = 0;

    v4i z4 = {0, 0, 0, 0};
    v8i z8 = {0, 0, 0, 0, 0, 0, 0, 0};
    __builtin_amdgcn_tensor_load_to_lds(g0, g1, z4, z4, z8, 0);
}

// =====================================================================
// Weight packing: conv weights W[O][CIN][3][3][3] (fp32) -> bf16 A-frags
// TAP-MAJOR layout for TDM: blk = (t*4 + mt)*KC + kc ; in-blk: lane*16+j
// =====================================================================
__global__ void pack_conv_w(const float* __restrict__ W, __bf16* __restrict__ P, int CIN) {
    const int KC = CIN / 32;
    const int total = 4 * 27 * KC * 512;
    for (int idx = blockIdx.x * blockDim.x + threadIdx.x; idx < total;
         idx += gridDim.x * blockDim.x) {
        int j    = idx & 15;
        int lane = (idx >> 4) & 31;
        int blk  = idx >> 9;
        int kc   = blk % KC;
        int mt   = (blk / KC) & 3;
        int t    = blk / (KC * 4);
        int o    = mt * 16 + (lane & 15);
        int c    = kc * 32 + kmapA(lane, j);
        int dz = t / 9, dy = (t / 3) % 3, dx = t % 3;
        float v = W[(((((long)o * CIN + c) * 3 + dz) * 3 + dy) * 3 + dx)];
        P[idx] = (__bf16)v;
    }
}

// Dense W[Mrows][K] (fp32, row-major) -> bf16 A-frags, blk = mt*KC + kc
__global__ void pack_dense_w(const float* __restrict__ W, __bf16* __restrict__ P,
                             int Mrows, int MT, int K) {
    const int KC = K / 32;
    const int total = MT * KC * 512;
    for (int idx = blockIdx.x * blockDim.x + threadIdx.x; idx < total;
         idx += gridDim.x * blockDim.x) {
        int j    = idx & 15;
        int lane = (idx >> 4) & 31;
        int blk  = idx >> 9;
        int kc   = blk % KC;
        int mt   = blk / KC;
        int m    = mt * 16 + (lane & 15);
        int k    = kc * 32 + kmapA(lane, j);
        float v  = (m < Mrows) ? W[(long)m * K + k] : 0.0f;
        P[idx] = (__bf16)v;
    }
}

// =====================================================================
// 3x3x3 conv as implicit GEMM. Block: 256 thr = 8 waves.
// Output tile: 64 channels x 32 voxels. Wave w -> (mt=w>>1, nt=w&1).
// Weights per tap TDM-prefetched (double-buffered) into LDS;
// K-loop: 27 taps x (CIN/32) chunks of V_WMMA_F32_16X16X32_BF16.
// =====================================================================
template <int CIN, bool IN_F32>
__global__ __launch_bounds__(256)
void conv3d_wmma(const void* __restrict__ inv,
                 const __bf16* __restrict__ wpack,
                 const float* __restrict__ bias,
                 __bf16* __restrict__ outp) {
    constexpr int KC = CIN / 32;
    constexpr int TAPB = 4 * KC * 512;                 // bf16 elems per tap
    __shared__ __align__(32) __bf16 slab[9 * SLAB_W * CIN];
    __shared__ __align__(32) __bf16 abuf[2][TAPB];

    const int xbase = blockIdx.x * TILE_X;
    const int y = blockIdx.y, z = blockIdx.z;
    const int tid = threadIdx.x;
    const int w    = tid >> 5;
    const int lane = tid & 31;

    // prefetch tap 0 weights via Tensor Data Mover (wave 0 issues)
    if (w == 0) {
        tdm_load_1d(wpack, &abuf[0][0], TAPB * 2, (unsigned long long)27 * TAPB * 2);
    }

    // ---- stage input slab into LDS, channel-innermost, zero-padded ----
    const int total = 9 * SLAB_W * CIN;
    for (int s = tid; s < total; s += 256) {
        int c   = s % CIN;
        int xx  = (s / CIN) % SLAB_W;
        int row = s / (CIN * SLAB_W);
        int gz = z + row / 3 - 1;
        int gy = y + row % 3 - 1;
        int gx = xbase + xx - 1;
        __bf16 v = (__bf16)0.0f;
        if ((unsigned)gz < (unsigned)D3 && (unsigned)gy < (unsigned)D3 &&
            (unsigned)gx < (unsigned)D3) {
            long gi = (((long)c * D3 + gz) * D3 + gy) * D3 + gx;
            if (IN_F32) v = (__bf16)((const float*)inv)[gi];
            else        v = ((const __bf16*)inv)[gi];
        }
        slab[s] = v;
    }

    const int mt   = w >> 1;
    const int nt   = w & 1;
    const int ncol = lane & 15;
    const int kbB  = (lane < 16) ? 0 : 16;   // B-fragment K half
    const int xl   = ncol + nt * 16;

    v8f acc = {};
    for (int t = 0; t < 27; ++t) {
        const int cur = t & 1;
        if (w == 0) __builtin_amdgcn_s_wait_tensorcnt(0);  // abuf[cur] landed
        __syncthreads();                                   // publish LDS (slab on t=0 too)
        if (w == 0 && t + 1 < 27) {                        // stream next tap
            tdm_load_1d(wpack + (long)(t + 1) * TAPB, &abuf[1 - cur][0],
                        TAPB * 2, (unsigned long long)27 * TAPB * 2);
        }
        const int row = t / 3, dx = t % 3;
        const __bf16* bbase = &slab[(row * SLAB_W + xl + dx) * CIN + kbB];
        const v16bf* av = (const v16bf*)&abuf[cur][0];
#pragma unroll
        for (int kc = 0; kc < KC; ++kc) {
            v16bf a = av[(mt * KC + kc) * 32 + lane];
            v16bf b = *(const v16bf*)(bbase + kc * 32);
            acc = __builtin_amdgcn_wmma_f32_16x16x32_bf16(
                false, a, false, b, (short)0, acc, false, false);
        }
        __syncthreads();   // all waves done reading abuf[cur] before overwrite
    }

    // ---- epilogue: bias + ReLU, store bf16 channel-major ----
    const int gx = xbase + xl;
    if (gx < D3) {
        const int m0 = mt * 16 + ((lane < 16) ? 0 : 8);
        const long vbase = ((long)z * D3 + y) * D3 + gx;
#pragma unroll
        for (int r = 0; r < 8; ++r) {
            int o = m0 + r;
            float val = fmaxf(acc[r] + bias[o], 0.0f);
            outp[(long)o * VOX + vbase] = (__bf16)val;
        }
    }
}

// =====================================================================
// Fused gather + 4-layer MLP. Block: 256 thr = 8 waves, 64 points.
// Activations ping-pong in LDS (bf16, channel-innermost per point).
// =====================================================================
template <int MT, int KC, bool RELU>
__device__ inline void mlp_layer(const __bf16* __restrict__ act_in, int kstride,
                                 __bf16* __restrict__ act_out, int ostride,
                                 const __bf16* __restrict__ wp,
                                 const float* __restrict__ bias, int tid) {
    const int w    = tid >> 5;
    const int lane = tid & 31;
    const int ncol = lane & 15;
    const int kbB  = (lane < 16) ? 0 : 16;
    const v16bf* wv = (const v16bf*)wp;
    for (int ot = w; ot < MT * 4; ot += 8) {
        const int mt = ot >> 2, nt = ot & 3;
        v8f acc = {};
        const __bf16* bbase = act_in + (nt * 16 + ncol) * kstride + kbB;
#pragma unroll
        for (int kc = 0; kc < KC; ++kc) {
            v16bf a = wv[(long)(mt * KC + kc) * 32 + lane];
            v16bf b = *(const v16bf*)(bbase + kc * 32);
            acc = __builtin_amdgcn_wmma_f32_16x16x32_bf16(
                false, a, false, b, (short)0, acc, false, false);
        }
        const int m0 = mt * 16 + ((lane < 16) ? 0 : 8);
        __bf16* obase = act_out + (nt * 16 + ncol) * ostride + m0;
#pragma unroll
        for (int r = 0; r < 8; ++r) {
            float v = acc[r] + bias[m0 + r];
            if (RELU) v = fmaxf(v, 0.0f);
            obase[r] = (__bf16)v;
        }
    }
}

__global__ __launch_bounds__(256)
void mlp_wmma(const __bf16* __restrict__ f2,
              const int* __restrict__ c0, const int* __restrict__ c1,
              const int* __restrict__ c2,
              const __bf16* __restrict__ wp1, const float* __restrict__ b1,
              const __bf16* __restrict__ wp2, const float* __restrict__ b2,
              const __bf16* __restrict__ wp3, const float* __restrict__ b3,
              const __bf16* __restrict__ wp4, const float* __restrict__ b4,
              float* __restrict__ outp) {
    __shared__ __align__(32) __bf16 bufA[64 * 256];   // g (64ch) then h2 (256ch)
    __shared__ __align__(32) __bf16 bufB[64 * 256];   // h1 (256ch) then h3 (64ch)
    int* sBase = (int*)bufB;  // alias: only used during gather, before h1 write

    const int tid = threadIdx.x;
    const int p0  = blockIdx.x * 64;

    if (tid < 64) {
        int p = p0 + tid;
        sBase[tid] = (c0[p] * D3 + c1[p]) * D3 + c2[p];
    }
    __syncthreads();

    // gather 64 channels per point into bufA (stride 64)
    {
        const int i = tid & 63, grp = tid >> 6;
        const int base = sBase[i];
#pragma unroll
        for (int j = 0; j < 16; ++j) {
            int c = grp * 16 + j;
            bufA[i * 64 + c] = f2[(long)c * VOX + base];
        }
    }
    __syncthreads();

    mlp_layer<16, 2, true >(bufA, 64,  bufB, 256, wp1, b1, tid);  // 64 -> 256
    __syncthreads();
    mlp_layer<16, 8, true >(bufB, 256, bufA, 256, wp2, b2, tid);  // 256 -> 256
    __syncthreads();
    mlp_layer<4,  8, false>(bufA, 256, bufB, 64,  wp3, b3, tid);  // 256 -> 64
    __syncthreads();

    // final layer: 64 -> 6 (M padded to 16), fp32 output
    {
        const int w    = tid >> 5;
        const int lane = tid & 31;
        const int ncol = lane & 15;
        const int kbB  = (lane < 16) ? 0 : 16;
        if (w < 4) {
            const int nt = w;
            v8f acc = {};
            const __bf16* bbase = bufB + (nt * 16 + ncol) * 64 + kbB;
            const v16bf* wv = (const v16bf*)wp4;
#pragma unroll
            for (int kc = 0; kc < 2; ++kc) {
                v16bf a = wv[kc * 32 + lane];
                v16bf b = *(const v16bf*)(bbase + kc * 32);
                acc = __builtin_amdgcn_wmma_f32_16x16x32_bf16(
                    false, a, false, b, (short)0, acc, false, false);
            }
            if (lane < 16) {  // rows 0..7 hold out channels 0..5
                int p = p0 + nt * 16 + ncol;
#pragma unroll
                for (int r = 0; r < 6; ++r)
                    outp[(long)r * NPTS + p] = acc[r] + b4[r];
            }
        }
    }
}

// =====================================================================
// host launcher
// =====================================================================
extern "C" void kernel_launch(void* const* d_in, const int* in_sizes, int n_in,
                              void* d_out, int out_size, void* d_ws, size_t ws_size,
                              hipStream_t stream) {
    const float* img    = (const float*)d_in[0];
    const int*   c0     = (const int*)d_in[1];
    const int*   c1     = (const int*)d_in[2];
    const int*   c2     = (const int*)d_in[3];
    const float* w_enc1 = (const float*)d_in[4];
    const float* b_enc1 = (const float*)d_in[5];
    const float* w_enc2 = (const float*)d_in[6];
    const float* b_enc2 = (const float*)d_in[7];
    const float* w_p1   = (const float*)d_in[8];
    const float* b_p1   = (const float*)d_in[9];
    const float* w_p2   = (const float*)d_in[10];
    const float* b_p2   = (const float*)d_in[11];
    const float* w_p3   = (const float*)d_in[12];
    const float* b_p3   = (const float*)d_in[13];
    const float* w_p4   = (const float*)d_in[14];
    const float* b_p4   = (const float*)d_in[15];
    float* out = (float*)d_out;

    char* ws = (char*)d_ws;
    size_t off = 0;
    auto take = [&](size_t n) -> char* {
        char* p = ws + off;
        off = (off + n + 255) & ~(size_t)255;
        return p;
    };
    __bf16* f1  = (__bf16*)take((size_t)64 * VOX * 2);       // conv1 out (bf16)
    __bf16* f2  = (__bf16*)take((size_t)64 * VOX * 2);       // conv2 out (bf16)
    __bf16* w1p = (__bf16*)take((size_t)4 * 27 * 1 * 512 * 2);
    __bf16* w2p = (__bf16*)take((size_t)4 * 27 * 2 * 512 * 2);
    __bf16* p1  = (__bf16*)take((size_t)16 * 2 * 512 * 2);
    __bf16* p2  = (__bf16*)take((size_t)16 * 8 * 512 * 2);
    __bf16* p3  = (__bf16*)take((size_t)4 * 8 * 512 * 2);
    __bf16* p4  = (__bf16*)take((size_t)1 * 2 * 512 * 2);

    // pack weights into CDNA5 A-fragment layout (bf16)
    pack_conv_w<<<64, 256, 0, stream>>>(w_enc1, w1p, 32);
    pack_conv_w<<<128, 256, 0, stream>>>(w_enc2, w2p, 64);
    pack_dense_w<<<32, 256, 0, stream>>>(w_p1, p1, 256, 16, 64);
    pack_dense_w<<<64, 256, 0, stream>>>(w_p2, p2, 256, 16, 256);
    pack_dense_w<<<32, 256, 0, stream>>>(w_p3, p3, 64, 4, 256);
    pack_dense_w<<<4, 256, 0, stream>>>(w_p4, p4, 6, 1, 64);

    dim3 cgrid((D3 + TILE_X - 1) / TILE_X, D3, D3);  // (3, 80, 80)
    conv3d_wmma<32, true ><<<cgrid, 256, 0, stream>>>(img, w1p, b_enc1, f1);
    conv3d_wmma<64, false><<<cgrid, 256, 0, stream>>>(f1, w2p, b_enc2, f2);

    mlp_wmma<<<NPTS / 64, 256, 0, stream>>>(f2, c0, c1, c2,
                                            p1, b_p1, p2, b_p2,
                                            p3, b_p3, p4, b_p4, out);
}
